// DeformableFeatureAggregationLite_16097537425515
// MI455X (gfx1250) — compile-verified, matrix operand order checked by reading, not run on hardware
//
#include <hip/hip_runtime.h>
#include <hip/hip_bf16.h>
#include <math.h>

// ---------------- problem constants ----------------
#define NB   2
#define NA   900
#define BA   (NB * NA)          // 1800 rows
#define ED   256
#define NG   8
#define GD   32
#define NCAM 6
#define NL   4
#define NP   7
#define NW   (NG * NCAM * NL * NP)   // 1344 fc outputs per row
#define NPTS (NCAM * NL * NP)        // 168 softmax entries per group

// WMMA types (probe-confirmed signatures for gfx1250 / ROCm 7.2)
typedef __attribute__((ext_vector_type(16))) __bf16 v16bf;
typedef __attribute__((ext_vector_type(8)))  float  v8f;

union ABFrag { uint4 u4[2]; v16bf v; };

__device__ __forceinline__ unsigned short f2bf(float f) {
    unsigned int u = __float_as_uint(f);
    u += 0x7FFFu + ((u >> 16) & 1u);     // round-to-nearest-even
    return (unsigned short)(u >> 16);
}

// FIX_SCALE keypoint offsets
__device__ const float kFixScale[NP][3] = {
    {0.f, 0.f, 0.f}, {0.45f, 0.f, 0.f}, {-0.45f, 0.f, 0.f},
    {0.f, 0.45f, 0.f}, {0.f, -0.45f, 0.f}, {0.f, 0.f, 0.45f},
    {0.f, 0.f, -0.45f}};

// ---------------- fp32 -> bf16 conversion kernels ----------------
__global__ __launch_bounds__(256) void convert_bf16_kernel(
    const float* __restrict__ src, unsigned short* __restrict__ dst, int n) {
    for (int i = blockIdx.x * blockDim.x + threadIdx.x; i < n;
         i += gridDim.x * blockDim.x)
        dst[i] = f2bf(src[i]);
}

__global__ __launch_bounds__(256) void add_convert_bf16_kernel(
    const float* __restrict__ a, const float* __restrict__ b,
    unsigned short* __restrict__ dst, int n) {
    for (int i = blockIdx.x * blockDim.x + threadIdx.x; i < n;
         i += gridDim.x * blockDim.x)
        dst[i] = f2bf(a[i] + b[i]);
}

// ---------------- bf16 WMMA GEMM: C[M,N] = A[M,K] * B[N,K]^T + bias[N] ------
// A row-major [M,K] bf16; B stored row-major [N,K] bf16 (i.e. already B^T).
// 8 waves per block; wave grid 2(M) x 4(N) -> 32x64 macro tile.
// Fragment layout per CDNA5 ISA 7.12.2 (16-bit A/B, wave32):
//   lanes 0-15  : K in {kb..kb+7} U {kb+16..kb+23}
//   lanes 16-31 : K in {kb+8..kb+15} U {kb+24..kb+31}
__global__ __launch_bounds__(256) void wmma_gemm_bias_kernel(
    const unsigned short* __restrict__ Abf,
    const unsigned short* __restrict__ Bbf,
    const float* __restrict__ bias,
    float* __restrict__ C,
    int M, int N, int K)
{
    const int lane    = threadIdx.x & 31;
    const int wave    = threadIdx.x >> 5;
    const int wm      = wave & 1;
    const int wn      = wave >> 1;
    const int m0      = blockIdx.x * 32 + wm * 16;
    const int n0      = blockIdx.y * 64 + wn * 16;
    const int l15     = lane & 15;
    const int halfSel = lane >> 4;           // 0 or 1 (K-phase of this lane)

    int arow = m0 + l15;                     // clamp M edge on loads
    if (arow > M - 1) arow = M - 1;
    const int ncol = n0 + l15;               // N is a multiple of 64 -> in range

    const unsigned short* Aptr = Abf + (size_t)arow * K + halfSel * 8;
    const unsigned short* Bptr = Bbf + (size_t)ncol * K + halfSel * 8;

    v8f acc = {};
    #pragma unroll
    for (int k0 = 0; k0 < 256; k0 += 32) {   // K == 256 for both GEMMs
        ABFrag a, b;
        a.u4[0] = *reinterpret_cast<const uint4*>(Aptr + k0);
        a.u4[1] = *reinterpret_cast<const uint4*>(Aptr + k0 + 16);
        b.u4[0] = *reinterpret_cast<const uint4*>(Bptr + k0);
        b.u4[1] = *reinterpret_cast<const uint4*>(Bptr + k0 + 16);
        acc = __builtin_amdgcn_wmma_f32_16x16x32_bf16(
            false, a.v, false, b.v, (short)0, acc, false, false);
    }

    // C/D layout: VGPR r -> row m0 + r + 8*halfSel, col n0 + (lane&15)
    const float bv = bias[ncol];
    #pragma unroll
    for (int r = 0; r < 8; ++r) {
        const int row = m0 + r + (halfSel ? 8 : 0);
        if (row < M) C[(size_t)row * N + ncol] = acc[r] + bv;
    }
}

// ---------------- per-(row,group) softmax over 168 entries ----------------
// logits layout per row: [p(168), g(8)], softmax over p for each g.
// One block per row; wave g handles group g (wave32, warpSize == 32).
__global__ __launch_bounds__(256) void softmax_kernel(float* __restrict__ wts) {
    const int idx  = blockIdx.x;              // row in [0, BA)
    const int lane = threadIdx.x & 31;
    const int g    = threadIdx.x >> 5;        // 8 waves -> 8 groups
    float* base = wts + (size_t)idx * NW;

    float vals[6];
    float mx = -INFINITY;
    #pragma unroll
    for (int i = 0; i < 6; ++i) {
        const int p = lane + i * 32;
        vals[i] = (p < NPTS) ? base[p * NG + g] : -INFINITY;
        mx = fmaxf(mx, vals[i]);
    }
    #pragma unroll
    for (int off = 16; off >= 1; off >>= 1)
        mx = fmaxf(mx, __shfl_xor(mx, off, 32));

    float sum = 0.f;
    #pragma unroll
    for (int i = 0; i < 6; ++i) {
        vals[i] = __expf(vals[i] - mx);       // exp(-inf) == 0 for pad lanes
        sum += vals[i];
    }
    #pragma unroll
    for (int off = 16; off >= 1; off >>= 1)
        sum += __shfl_xor(sum, off, 32);
    const float inv = 1.f / sum;

    #pragma unroll
    for (int i = 0; i < 6; ++i) {
        const int p = lane + i * 32;
        if (p < NPTS) base[p * NG + g] = vals[i] * inv;
    }
}

// ---------------- keypoint projection + bilinear sampling + fusion --------
// One block per (b,a) anchor, 256 threads = one channel per thread.
// Geometry (42 projected points) and 1344 attention weights staged in LDS.
__global__ __launch_bounds__(256) void sample_fuse_kernel(
    const float* __restrict__ anchor,    // [B,A,8]
    const float* __restrict__ proj,      // [B,6,4,4]
    const float* __restrict__ wh,        // [B,6,2]
    const float* __restrict__ f0, const float* __restrict__ f1,
    const float* __restrict__ f2, const float* __restrict__ f3,
    const float* __restrict__ wts,       // [BA, 1344] softmaxed
    unsigned short* __restrict__ fusedbf)// [BA, 256] bf16
{
    __shared__ float s_xy[NCAM][NP][2];  // normalized grid coords
    __shared__ float s_w[NW];            // attention weights for this anchor

    const int idx = blockIdx.x;          // 0..BA-1
    const int b   = idx / NA;
    const int tid = threadIdx.x;

    if (tid < NCAM * NP) {
        const int c = tid / NP, p = tid % NP;
        const float* an = anchor + (size_t)idx * 8;
        const float cx = an[0], cy = an[1], cz = an[2];
        const float sx = __expf(an[3]), sy = __expf(an[4]), sz = __expf(an[5]);
        const float sn = an[6], cs = an[7];
        const float kx = kFixScale[p][0] * sx;
        const float ky = kFixScale[p][1] * sy;
        const float kz = kFixScale[p][2] * sz;
        // rot = [[cos,-sin,0],[sin,cos,0],[0,0,1]] then translate
        const float wx = cs * kx - sn * ky + cx;
        const float wy = sn * kx + cs * ky + cy;
        const float wz = kz + cz;
        const float* Mt = proj + ((size_t)b * NCAM + c) * 16;
        const float px = Mt[0] * wx + Mt[1] * wy + Mt[2]  * wz + Mt[3];
        const float py = Mt[4] * wx + Mt[5] * wy + Mt[6]  * wz + Mt[7];
        const float pz = Mt[8] * wx + Mt[9] * wy + Mt[10] * wz + Mt[11];
        const float zz = fmaxf(pz, 1e-5f);
        const float iw = fmaxf(wh[((size_t)b * NCAM + c) * 2 + 0], 1e-5f);
        const float ih = fmaxf(wh[((size_t)b * NCAM + c) * 2 + 1], 1e-5f);
        s_xy[c][p][0] = (px / zz) / iw * 2.f - 1.f;
        s_xy[c][p][1] = (py / zz) / ih * 2.f - 1.f;
    }
    for (int j = tid; j < NW; j += 256) s_w[j] = wts[(size_t)idx * NW + j];
    __syncthreads();

    const int d = tid;                    // channel
    const int g = d >> 5;                 // group = d / GD
    const int Hs[NL] = {64, 32, 16, 8};
    const int Ws[NL] = {176, 88, 44, 22};
    const float* fl[NL] = {f0, f1, f2, f3};

    float acc = 0.f;
    for (int c = 0; c < NCAM; ++c) {
        #pragma unroll
        for (int l = 0; l < NL; ++l) {
            const int H = Hs[l], W = Ws[l];
            const float* fp = fl[l] +
                (((size_t)b * NCAM + c) * ED + d) * (size_t)(H * W);
            #pragma unroll
            for (int p = 0; p < NP; ++p) {
                // align_corners=False mapping
                const float gx = (s_xy[c][p][0] + 1.f) * (W * 0.5f) - 0.5f;
                const float gy = (s_xy[c][p][1] + 1.f) * (H * 0.5f) - 0.5f;
                const float fx = floorf(gx), fy = floorf(gy);
                const int x0 = (int)fx, y0 = (int)fy;
                const float wx1 = gx - fx, wy1 = gy - fy;
                const float wx0 = 1.f - wx1, wy0 = 1.f - wy1;
                float v = 0.f;
                const bool xin0 = (x0 >= 0) & (x0 < W);
                const bool xin1 = (x0 + 1 >= 0) & (x0 + 1 < W);
                const bool yin0 = (y0 >= 0) & (y0 < H);
                const bool yin1 = (y0 + 1 >= 0) & (y0 + 1 < H);
                if (xin0 & yin0) v += wx0 * wy0 * fp[y0 * W + x0];
                if (xin1 & yin0) v += wx1 * wy0 * fp[y0 * W + x0 + 1];
                if (xin0 & yin1) v += wx0 * wy1 * fp[(y0 + 1) * W + x0];
                if (xin1 & yin1) v += wx1 * wy1 * fp[(y0 + 1) * W + x0 + 1];
                acc += s_w[((c * NL + l) * NP + p) * NG + g] * v;
            }
        }
    }
    fusedbf[(size_t)idx * ED + d] = f2bf(acc);
}

// ---------------- host-side orchestration ----------------
extern "C" void kernel_launch(void* const* d_in, const int* in_sizes, int n_in,
                              void* d_out, int out_size, void* d_ws, size_t ws_size,
                              hipStream_t stream) {
    (void)in_sizes; (void)n_in; (void)out_size; (void)ws_size;
    const float* inst   = (const float*)d_in[0];
    const float* anchor = (const float*)d_in[1];
    const float* embed  = (const float*)d_in[2];
    const float* f0     = (const float*)d_in[3];
    const float* f1     = (const float*)d_in[4];
    const float* f2     = (const float*)d_in[5];
    const float* f3     = (const float*)d_in[6];
    const float* proj   = (const float*)d_in[7];
    const float* wh     = (const float*)d_in[8];
    const float* w_fc   = (const float*)d_in[9];
    const float* b_fc   = (const float*)d_in[10];
    const float* w_out  = (const float*)d_in[11];
    const float* b_out  = (const float*)d_in[12];
    float* out = (float*)d_out;

    // workspace sub-allocation (256B aligned)
    char* ws = (char*)d_ws;
    size_t off = 0;
    auto suballoc = [&](size_t bytes) {
        void* p = ws + off;
        off = (off + bytes + 255) & ~(size_t)255;
        return p;
    };
    unsigned short* featbf  = (unsigned short*)suballoc((size_t)BA * ED * 2);
    unsigned short* wfcbf   = (unsigned short*)suballoc((size_t)NW * ED * 2);
    unsigned short* woutbf  = (unsigned short*)suballoc((size_t)ED * ED * 2);
    float*          weights = (float*)        suballoc((size_t)BA * NW * 4);
    unsigned short* fusedbf = (unsigned short*)suballoc((size_t)BA * ED * 2);

    // 1. fp32 -> bf16 conversions
    {
        int n = BA * ED;
        add_convert_bf16_kernel<<<(n + 255) / 256, 256, 0, stream>>>(
            inst, embed, featbf, n);
        n = NW * ED;
        convert_bf16_kernel<<<(n + 255) / 256, 256, 0, stream>>>(w_fc, wfcbf, n);
        n = ED * ED;
        convert_bf16_kernel<<<(n + 255) / 256, 256, 0, stream>>>(w_out, woutbf, n);
    }

    // 2. logits GEMM: [1800,256] x [1344,256]^T + b_fc
    {
        dim3 grid((BA + 31) / 32, NW / 64);
        wmma_gemm_bias_kernel<<<grid, 256, 0, stream>>>(
            featbf, wfcbf, b_fc, weights, BA, NW, ED);
    }

    // 3. group softmax (in place)
    softmax_kernel<<<BA, 256, 0, stream>>>(weights);

    // 4. project + bilinear sample + weighted fusion
    sample_fuse_kernel<<<BA, 256, 0, stream>>>(
        anchor, proj, wh, f0, f1, f2, f3, weights, fusedbf);

    // 5. output GEMM: [1800,256] x [256,256]^T + b_out -> d_out
    {
        dim3 grid((BA + 31) / 32, ED / 64);
        wmma_gemm_bias_kernel<<<grid, 256, 0, stream>>>(
            fusedbf, woutbf, b_out, out, BA, ED, ED);
    }
}
